// AttentionMechanism_55207509622811
// MI455X (gfx1250) — compile-verified, matrix operand order checked by reading, not run on hardware
//
#include <hip/hip_runtime.h>

// Problem constants
#define CD   256
#define BB   64
#define HID  512
#define NBETA 512
#define HW   4096

typedef __attribute__((ext_vector_type(16))) __bf16 v16bf;
typedef __attribute__((ext_vector_type(8)))  float  v8f;

union Frag { uint4 u[2]; v16bf v; };

__device__ __forceinline__ unsigned short f2bf(float x) {
    unsigned u = __float_as_uint(x);
    unsigned r = 0x7FFFu + ((u >> 16) & 1u);   // round-to-nearest-even
    return (unsigned short)((u + r) >> 16);
}

// ---------------- Kernel 1: W_w fp32 -> bf16 (row-major 512x256) --------------
__global__ __launch_bounds__(256)
void convw_kernel(const float* __restrict__ W_w, unsigned short* __restrict__ Wbf) {
    int idx = blockIdx.x * 256 + threadIdx.x;     // grid 512 -> exactly 131072
    Wbf[idx] = f2bf(W_w[idx]);
}

// ---------------- Kernel 2: G[j,b] = h_t[b]·W_h_w[j] + W_h_b[j] + W_b[j] ------
__global__ __launch_bounds__(64)
void bias_kernel(const float* __restrict__ h_t, const float* __restrict__ W_h_w,
                 const float* __restrict__ W_h_b, const float* __restrict__ W_b,
                 float* __restrict__ G) {
    int j = blockIdx.x;           // 0..511
    int b = threadIdx.x;          // 0..63
    const float* wr = W_h_w + (size_t)j * HID;
    const float* hr = h_t  + (size_t)b * HID;
    float acc = 0.f;
    #pragma unroll 8
    for (int h = 0; h < HID; ++h) acc = fmaf(wr[h], hr[h], acc);
    G[j * BB + b] = acc + W_h_b[j] + W_b[j];
}

// ---------------- Kernel 3: WMMA attention scores -----------------------------
// Per position p: T = W_w(512x256) @ V[p](256x64); E[p,b] = sum_j beta_w[j]*tanh(T[j,b]+G[j,b])
#define ROWB   528                 // padded LDS row: 264 bf16 = 528 B (33*16 -> conflict-free b128)
#define W_BYTES (512 * ROWB)
#define V_BYTES (64 * ROWB)
#define E_OFF   (W_BYTES + V_BYTES)
#define LDS_TOTAL (E_OFF + 64 * 4)

__global__ __launch_bounds__(256, 1)
void attn_scores_kernel(const float* __restrict__ Vg,
                        const unsigned short* __restrict__ Wbf,
                        const float* __restrict__ G,
                        const float* __restrict__ beta_w,
                        float* __restrict__ Eg,
                        int pos_per_wg) {
    extern __shared__ char lds[];
    char*  Wl = lds;
    char*  Vl = lds + W_BYTES;
    float* El = (float*)(lds + E_OFF);

    const int tid  = threadIdx.x;
    const int lane = tid & 31;
    const int wv   = tid >> 5;     // wave 0..7
    const int mw   = wv >> 1;      // beta block of 128
    const int nw   = wv & 1;       // batch block of 32
    const int lrow = lane & 15;
    const int lhi  = lane >> 4;    // 0/1
    const int kb   = lhi * 8;      // K sub-offset within 32-wide tile

    // Stage all of W_w (bf16) into LDS once per workgroup (reused for 16 positions)
    {
        const uint4* gw = (const uint4*)Wbf;   // 16384 x uint4
        #pragma unroll 4
        for (int i = 0; i < 64; ++i) {
            int idx = tid + i * 256;
            int j  = idx >> 5;
            int ch = idx & 31;
            *(uint4*)(Wl + j * ROWB + ch * 16) = gw[idx];
        }
    }

    for (int it = 0; it < pos_per_wg; ++it) {
        int p = blockIdx.x * pos_per_wg + it;
        __syncthreads();           // protect V_l / E_l from previous iteration readers
        // Stage V[p] (256x64 fp32, b fastest -> coalesced) into LDS as bf16 [b][c]
        {
            const float* vp = Vg + (size_t)p * (CD * BB);
            int b = tid & 63;
            int cbase = tid >> 6;
            #pragma unroll 4
            for (int i = 0; i < 64; ++i) {
                int c = cbase + i * 4;
                *(unsigned short*)(Vl + b * ROWB + c * 2) = f2bf(vp[c * BB + b]);
            }
        }
        if (tid < 64) El[tid] = 0.f;
        __syncthreads();

        v8f acc[8][2];
        {
            v8f z = {0.f,0.f,0.f,0.f,0.f,0.f,0.f,0.f};
            #pragma unroll
            for (int mt = 0; mt < 8; ++mt) { acc[mt][0] = z; acc[mt][1] = z; }
        }

        for (int ks = 0; ks < 8; ++ks) {       // K = 256, 32 per step
            int c0 = ks * 32;
            Frag bf0, bf1;
            {
                int b0 = nw * 32 + lrow;
                const char* base0 = Vl + b0 * ROWB + (c0 + kb) * 2;
                bf0.u[0] = *(const uint4*)(base0);
                bf0.u[1] = *(const uint4*)(base0 + 32);
                const char* base1 = Vl + (b0 + 16) * ROWB + (c0 + kb) * 2;
                bf1.u[0] = *(const uint4*)(base1);
                bf1.u[1] = *(const uint4*)(base1 + 32);
            }
            #pragma unroll
            for (int mt = 0; mt < 8; ++mt) {
                Frag af;
                int j = mw * 128 + mt * 16 + lrow;
                const char* ab = Wl + j * ROWB + (c0 + kb) * 2;
                af.u[0] = *(const uint4*)(ab);
                af.u[1] = *(const uint4*)(ab + 32);
                acc[mt][0] = __builtin_amdgcn_wmma_f32_16x16x32_bf16(
                    false, af.v, false, bf0.v, (short)0, acc[mt][0], false, false);
                acc[mt][1] = __builtin_amdgcn_wmma_f32_16x16x32_bf16(
                    false, af.v, false, bf1.v, (short)0, acc[mt][1], false, false);
            }
        }

        // E contribution: C/D layout -> lane l, reg r holds (j = base+8*lhi+r, b = base+(l&15))
        float s0 = 0.f, s1 = 0.f;
        int b0 = nw * 32 + lrow;
        #pragma unroll
        for (int mt = 0; mt < 8; ++mt) {
            #pragma unroll
            for (int r = 0; r < 8; ++r) {
                int j = mw * 128 + mt * 16 + lhi * 8 + r;
                float bw = beta_w[j];
                float u0 = acc[mt][0][r] + G[j * BB + b0];
                float u1 = acc[mt][1][r] + G[j * BB + b0 + 16];
                s0 += bw * tanhf(u0);
                s1 += bw * tanhf(u1);
            }
        }
        // combine lane halves (same b, complementary j) via ds_swizzle xor-16
        s0 += __int_as_float(__builtin_amdgcn_ds_swizzle(__float_as_int(s0), 0x401F));
        s1 += __int_as_float(__builtin_amdgcn_ds_swizzle(__float_as_int(s1), 0x401F));
        if (lhi == 0) {
            atomicAdd(&El[b0], s0);        // cross-wave (mw) reduce in LDS
            atomicAdd(&El[b0 + 16], s1);
        }
        __syncthreads();
        if (tid < 64) Eg[(size_t)p * BB + tid] = El[tid];
    }
}

// ---------------- Kernel 4: softmax over positions, per batch -----------------
__global__ __launch_bounds__(256)
void softmax_kernel(const float* __restrict__ Eg, float* __restrict__ Ag) {
    __shared__ float red[256];
    int b = blockIdx.x, tid = threadIdx.x;
    float m = -3.402823466e38f;
    for (int p = tid; p < HW; p += 256) m = fmaxf(m, Eg[p * BB + b]);
    red[tid] = m; __syncthreads();
    for (int s = 128; s > 0; s >>= 1) { if (tid < s) red[tid] = fmaxf(red[tid], red[tid + s]); __syncthreads(); }
    float M = red[0]; __syncthreads();
    float S = 0.f;
    for (int p = tid; p < HW; p += 256) S += __expf(Eg[p * BB + b] - M);
    red[tid] = S; __syncthreads();
    for (int s = 128; s > 0; s >>= 1) { if (tid < s) red[tid] += red[tid + s]; __syncthreads(); }
    float inv = 1.f / red[0];
    for (int p = tid; p < HW; p += 256) Ag[p * BB + b] = __expf(Eg[p * BB + b] - M) * inv;
}

// ---------------- Kernel 5: C[b,c] = sum_p A[p,b] * V[p,c,b] ------------------
__global__ __launch_bounds__(256)
void weighted_sum_kernel(const float* __restrict__ Vg, const float* __restrict__ Ag,
                         float* __restrict__ out) {
    __shared__ float red[256];
    int c = blockIdx.x, tid = threadIdx.x;
    int b = tid & 63, sl = tid >> 6;
    float acc = 0.f;
    for (int p = sl; p < HW; p += 4)
        acc = fmaf(Ag[p * BB + b], Vg[((size_t)p * CD + c) * BB + b], acc);
    red[tid] = acc; __syncthreads();
    if (tid < 64)
        out[b * CD + c] = red[tid] + red[tid + 64] + red[tid + 128] + red[tid + 192];
}

// ---------------- Launcher ----------------------------------------------------
extern "C" void kernel_launch(void* const* d_in, const int* in_sizes, int n_in,
                              void* d_out, int out_size, void* d_ws, size_t ws_size,
                              hipStream_t stream) {
    const float* V      = (const float*)d_in[0];
    const float* h_t    = (const float*)d_in[1];
    const float* W_h_w  = (const float*)d_in[2];
    const float* W_h_b  = (const float*)d_in[3];
    const float* W_w    = (const float*)d_in[4];
    const float* W_b    = (const float*)d_in[5];
    const float* beta_w = (const float*)d_in[6];
    // beta_b (d_in[7]) intentionally unused: softmax is shift-invariant.

    char* ws = (char*)d_ws;
    unsigned short* Wbf = (unsigned short*)ws;                       // 256 KiB bf16 W_w
    float* G  = (float*)(ws + 262144);                               // 128 KiB bias table
    float* Eg = (float*)(ws + 262144 + 131072);                      // 1 MiB scores [p][b]
    float* Ag = (float*)(ws + 262144 + 131072 + 1048576);            // 1 MiB softmax weights

    convw_kernel<<<512, 256, 0, stream>>>(W_w, Wbf);
    bias_kernel<<<NBETA, 64, 0, stream>>>(h_t, W_h_w, W_h_b, W_b, G);
    attn_scores_kernel<<<256, 256, LDS_TOTAL, stream>>>(V, Wbf, G, beta_w, Eg, 16);
    softmax_kernel<<<BB, 256, 0, stream>>>(Eg, Ag);
    weighted_sum_kernel<<<CD, 256, 0, stream>>>(V, Ag, (float*)d_out);
}
